// MultiHeadAttention_34102040330384
// MI455X (gfx1250) — matrix operand order repaired and checked by reading, not compiled, not run on hardware
//
#include <hip/hip_runtime.h>

// Multi-head attention forward for MI455X (gfx1250), wave32 + WMMA.
//
// Roofline: attn [2,16,2048,2048] fp32 (536 MB) dominates; it must be written
// (it is a reference output) but never needs to be re-read if scores+softmax+
// attn·V are fused. Pipeline (all on `stream`, in-order):
//   1) f32->f16 conversion of activations + weights
//   2) proj GEMMs: Q,K -> [B,H,S,64] f16 ; V -> transposed [B,H,64,S] f16
//   3) fused attention: per 16-row q-strip:
//        scores strip (16x2048) in registers via WMMA,
//        row softmax (shuffle + LDS reduce),
//        attn written once to d_out via nontemporal stores,
//        f16 strip kept in LDS (64 KB) and fed straight into attn·V WMMA,
//        cross-wave K reduction in LDS -> ctx f16 [B*S, D]
//   4) out GEMM: ctx · w_o^T + b_o -> fp32 d_out[0 .. B*S*D)
// All GEMMs use v_wmma_f32_16x16x32_f16 (fp32 accumulate). Workspace ~64 MB.

constexpr int B_  = 2;
constexpr int S_  = 2048;
constexpr int D_  = 1024;
constexpr int H_  = 16;
constexpr int DK_ = 64;
constexpr float SCALE_INV = 0.125f;   // 1/sqrt(64)

typedef __attribute__((ext_vector_type(16))) _Float16 v16h;
typedef __attribute__((ext_vector_type(8)))  _Float16 v8h;
typedef __attribute__((ext_vector_type(8)))  float    v8f;

static __device__ __forceinline__ v8f wmma16(v16h a, v16h b, v8f c) {
  // (neg_a, A, neg_b, B, c_mod, C, reuse_a, reuse_b)
  return __builtin_amdgcn_wmma_f32_16x16x32_f16(false, a, false, b, (short)0, c,
                                                false, false);
}

// A fragment: 16x32 (MxK) f16 tile from row-major A, leading dim lda (elements).
// ISA layout: lane L<16 holds row M=L, K = {0..7, 16..23}; lane L+16 -> {8..15, 24..31}.
static __device__ __forceinline__ v16h load_a_f16(const _Float16* __restrict__ A,
                                                  int lda, int row0, int k0) {
  const int lane = threadIdx.x & 31;
  const int half = lane >> 4;
  const _Float16* p = A + (size_t)(row0 + (lane & 15)) * lda + k0 + half * 8;
  v8h lo = *(const v8h*)p;
  v8h hi = *(const v8h*)(p + 16);
  v16h r;
#pragma unroll
  for (int i = 0; i < 8; ++i) { r[i] = lo[i]; r[i + 8] = hi[i]; }
  return r;
}

// B fragment: 32x16 (KxN) where B[k][n] = W[col0+n][k0+k], W row-major [N x K].
// ISA layout: lane n<16 holds column n, K = {0..7, 16..23}; lane n+16 -> {8..15, 24..31}.
static __device__ __forceinline__ v16h load_b_f16(const _Float16* __restrict__ W,
                                                  int ldw, int col0, int k0) {
  const int lane = threadIdx.x & 31;
  const int half = lane >> 4;
  const _Float16* p = W + (size_t)(col0 + (lane & 15)) * ldw + k0 + half * 8;
  v8h lo = *(const v8h*)p;
  v8h hi = *(const v8h*)(p + 16);
  v16h r;
#pragma unroll
  for (int i = 0; i < 8; ++i) { r[i] = lo[i]; r[i + 8] = hi[i]; }
  return r;
}

// ---------------------------------------------------------------- conversion
__global__ void mha_cvt_f32_f16(const float* __restrict__ src,
                                _Float16* __restrict__ dst, int n) {
  int i = blockIdx.x * blockDim.x + threadIdx.x;
  if (i < n) dst[i] = (_Float16)src[i];
}

// ---------------------------------------------------------------- projections
// C[m,n] = sum_d X[m,d] * W[n,d];  M=B*S=4096, N=D=1024, K=D=1024.
// mode 0: dst = head-split [B,H,S,64] f16 (Q, K)
// mode 1: dst = transposed [B,H,64,S] f16 (V)
__global__ __launch_bounds__(256) void mha_proj(const _Float16* __restrict__ X,
                                                const _Float16* __restrict__ W,
                                                _Float16* __restrict__ dst,
                                                int mode) {
  const int w  = threadIdx.x >> 5;
  const int wm = w >> 2, wn = w & 3;                 // 2x4 waves -> 64x128 block tile
  const int m0 = blockIdx.x * 64 + wm * 32;
  const int n0 = blockIdx.y * 128 + wn * 32;
  v8f acc[2][2] = {};
  for (int k0 = 0; k0 < D_; k0 += 32) {
    v16h a0 = load_a_f16(X, D_, m0,      k0);
    v16h a1 = load_a_f16(X, D_, m0 + 16, k0);
    v16h b0 = load_b_f16(W, D_, n0,      k0);
    v16h b1 = load_b_f16(W, D_, n0 + 16, k0);
    acc[0][0] = wmma16(a0, b0, acc[0][0]);
    acc[0][1] = wmma16(a0, b1, acc[0][1]);
    acc[1][0] = wmma16(a1, b0, acc[1][0]);
    acc[1][1] = wmma16(a1, b1, acc[1][1]);
  }
  const int lane = threadIdx.x & 31;
  const int cn = lane & 15, cm = (lane >> 4) * 8;
#pragma unroll
  for (int mi = 0; mi < 2; ++mi)
#pragma unroll
    for (int ni = 0; ni < 2; ++ni)
#pragma unroll
      for (int i = 0; i < 8; ++i) {
        const int m = m0 + mi * 16 + cm + i;
        const int n = n0 + ni * 16 + cn;
        const int b = m >> 11, s = m & (S_ - 1);
        const int h = n >> 6,  d = n & (DK_ - 1);
        size_t idx;
        if (mode == 0) idx = (((size_t)(b * H_ + h)) * S_ + s) * DK_ + d;
        else           idx = (((size_t)(b * H_ + h)) * DK_ + d) * S_ + s;
        dst[idx] = (_Float16)acc[mi][ni][i];
      }
}

// ---------------------------------------------------------------- fused attention
// One workgroup = one (b,h) x 16-row q-strip. 8 waves, wave w owns score
// columns [w*256, (w+1)*256).
// Phase 1: strip scores via WMMA (16 n-tiles/wave, accumulators stay in VGPRs)
// Phase 2/3: row max / exp-sum (wave shuffles + LDS tree across waves)
// Phase 4: attn = softmax written ONCE to global (nontemporal) + f16 strip in LDS
// Phase 5: ctx partial = strip_chunk(16x256) @ Vt via WMMA (A from LDS)
// Phase 6: cross-wave K reduction in LDS -> ctx16 global
__global__ __launch_bounds__(256) void mha_fused_attn(
    const _Float16* __restrict__ Qh, const _Float16* __restrict__ Kh,
    const _Float16* __restrict__ Vt, float* __restrict__ attn,
    _Float16* __restrict__ ctx16) {
  __shared__ __align__(16) _Float16 s_strip[16 * 2048];  // 64 KB, phase-multiplexed
  float* s_red  = (float*)s_strip;   // phases 2-3: [0..143] (before strip written)
  float* s_part = (float*)s_strip;   // phase 6: [8][16][64] (after strip consumed)

  const int bh = blockIdx.y;
  const int b  = bh >> 4, h = bh & (H_ - 1);
  const int m0 = blockIdx.x * 16;
  const _Float16* Q  = Qh + (size_t)bh * S_ * DK_;
  const _Float16* K  = Kh + (size_t)bh * S_ * DK_;
  const _Float16* Vb = Vt + (size_t)bh * DK_ * S_;
  float* arow = attn + (size_t)bh * S_ * S_ + (size_t)m0 * S_;

  const int w     = threadIdx.x >> 5;
  const int lane  = threadIdx.x & 31;
  const int cn    = lane & 15;
  const int cm    = (lane >> 4) * 8;
  const int ncol0 = w * 256;

  // ---- phase 1: scores strip (already scaled by 1/sqrt(dk))
  v16h a0 = load_a_f16(Q, DK_, m0, 0);
  v16h a1 = load_a_f16(Q, DK_, m0, 32);
  v8f acc[16];
#pragma unroll
  for (int nt = 0; nt < 16; ++nt) {
    v16h b0 = load_b_f16(K, DK_, ncol0 + nt * 16, 0);
    v16h b1 = load_b_f16(K, DK_, ncol0 + nt * 16, 32);
    v8f z = {};
    z = wmma16(a0, b0, z);
    acc[nt] = wmma16(a1, b1, z);
  }
#pragma unroll
  for (int nt = 0; nt < 16; ++nt)
#pragma unroll
    for (int i = 0; i < 8; ++i) acc[nt][i] *= SCALE_INV;

  // ---- phase 2: row max (per-lane -> 16 lanes via shuffle -> 8 waves via LDS)
  float rmax[8];
#pragma unroll
  for (int i = 0; i < 8; ++i) {
    float m = acc[0][i];
#pragma unroll
    for (int nt = 1; nt < 16; ++nt) m = fmaxf(m, acc[nt][i]);
#pragma unroll
    for (int mask = 1; mask <= 8; mask <<= 1)
      m = fmaxf(m, __shfl_xor(m, mask, 32));
    rmax[i] = m;
  }
  if (cn == 0) {
#pragma unroll
    for (int i = 0; i < 8; ++i) s_red[w * 16 + cm + i] = rmax[i];
  }
  __syncthreads();
  if (threadIdx.x < 16) {
    float g = s_red[threadIdx.x];
    for (int ww = 1; ww < 8; ++ww) g = fmaxf(g, s_red[ww * 16 + threadIdx.x]);
    s_red[128 + threadIdx.x] = g;
  }
  __syncthreads();
  float gmax[8];
#pragma unroll
  for (int i = 0; i < 8; ++i) gmax[i] = s_red[128 + cm + i];
  __syncthreads();

  // ---- phase 3: exp + row sum
  float rsum[8];
#pragma unroll
  for (int i = 0; i < 8; ++i) rsum[i] = 0.f;
#pragma unroll
  for (int nt = 0; nt < 16; ++nt)
#pragma unroll
    for (int i = 0; i < 8; ++i) {
      const float e = __expf(acc[nt][i] - gmax[i]);
      acc[nt][i] = e;
      rsum[i] += e;
    }
#pragma unroll
  for (int i = 0; i < 8; ++i) {
#pragma unroll
    for (int mask = 1; mask <= 8; mask <<= 1)
      rsum[i] += __shfl_xor(rsum[i], mask, 32);
  }
  if (cn == 0) {
#pragma unroll
    for (int i = 0; i < 8; ++i) s_red[w * 16 + cm + i] = rsum[i];
  }
  __syncthreads();
  if (threadIdx.x < 16) {
    float g = 0.f;
    for (int ww = 0; ww < 8; ++ww) g += s_red[ww * 16 + threadIdx.x];
    s_red[128 + threadIdx.x] = 1.0f / g;
  }
  __syncthreads();
  float rinv[8];
#pragma unroll
  for (int i = 0; i < 8; ++i) rinv[i] = s_red[128 + cm + i];
  __syncthreads();   // s_red reads complete; strip region now free

  // ---- phase 4: write attn (streaming, never re-read) + f16 strip in LDS
#pragma unroll
  for (int nt = 0; nt < 16; ++nt)
#pragma unroll
    for (int i = 0; i < 8; ++i) {
      const int row = cm + i;
      const int col = ncol0 + nt * 16 + cn;
      const float p = acc[nt][i] * rinv[i];
      __builtin_nontemporal_store(p, &arow[(size_t)row * S_ + col]);
      s_strip[row * 2048 + col] = (_Float16)p;
    }
  // wave w wrote and will read only columns [ncol0, ncol0+256): no barrier needed

  // ---- phase 5: ctx partial over this wave's K-chunk (A from LDS strip)
  v8f cacc[4] = {};
  for (int ks = 0; ks < 8; ++ks) {
    const int k0 = ncol0 + ks * 32;
    const int half = lane >> 4;
    const _Float16* ap = &s_strip[(lane & 15) * 2048 + k0 + half * 8];
    v8h lo = *(const v8h*)ap;
    v8h hi = *(const v8h*)(ap + 16);
    v16h af;
#pragma unroll
    for (int i = 0; i < 8; ++i) { af[i] = lo[i]; af[i + 8] = hi[i]; }
#pragma unroll
    for (int nt = 0; nt < 4; ++nt) {
      v16h bf = load_b_f16(Vb, S_, nt * 16, k0);
      cacc[nt] = wmma16(af, bf, cacc[nt]);
    }
  }
  __syncthreads();   // everyone done reading strip before it becomes s_part

  // ---- phase 6: cross-wave K reduction -> ctx16
#pragma unroll
  for (int nt = 0; nt < 4; ++nt)
#pragma unroll
    for (int i = 0; i < 8; ++i)
      s_part[(w * 16 + cm + i) * 64 + nt * 16 + cn] = cacc[nt][i];
  __syncthreads();
  for (int e = threadIdx.x; e < 16 * 64; e += 256) {
    const int r = e >> 6, c = e & 63;
    float s = 0.f;
#pragma unroll
    for (int ww = 0; ww < 8; ++ww) s += s_part[(ww * 16 + r) * 64 + c];
    ctx16[(size_t)(b * S_ + m0 + r) * D_ + h * DK_ + c] = (_Float16)s;
  }
}

// ---------------------------------------------------------------- output proj
// out[m,n] = sum_d ctx[m,d] * Wo[n,d] + bo[n]; fp32 out
__global__ __launch_bounds__(256) void mha_out(const _Float16* __restrict__ C16,
                                               const _Float16* __restrict__ Wo,
                                               const float* __restrict__ bo,
                                               float* __restrict__ out) {
  const int w  = threadIdx.x >> 5;
  const int wm = w >> 2, wn = w & 3;
  const int m0 = blockIdx.x * 64 + wm * 32;
  const int n0 = blockIdx.y * 128 + wn * 32;
  v8f acc[2][2] = {};
  for (int k0 = 0; k0 < D_; k0 += 32) {
    v16h a0 = load_a_f16(C16, D_, m0,      k0);
    v16h a1 = load_a_f16(C16, D_, m0 + 16, k0);
    v16h b0 = load_b_f16(Wo, D_, n0,      k0);
    v16h b1 = load_b_f16(Wo, D_, n0 + 16, k0);
    acc[0][0] = wmma16(a0, b0, acc[0][0]);
    acc[0][1] = wmma16(a0, b1, acc[0][1]);
    acc[1][0] = wmma16(a1, b0, acc[1][0]);
    acc[1][1] = wmma16(a1, b1, acc[1][1]);
  }
  const int lane = threadIdx.x & 31;
  const int cn = lane & 15, cm = (lane >> 4) * 8;
#pragma unroll
  for (int mi = 0; mi < 2; ++mi)
#pragma unroll
    for (int ni = 0; ni < 2; ++ni)
#pragma unroll
      for (int i = 0; i < 8; ++i) {
        const int m = m0 + mi * 16 + cm + i;
        const int n = n0 + ni * 16 + cn;
        out[(size_t)m * D_ + n] = acc[mi][ni][i] + bo[n];
      }
}

// ---------------------------------------------------------------- launch
extern "C" void kernel_launch(void* const* d_in, const int* in_sizes, int n_in,
                              void* d_out, int out_size, void* d_ws, size_t ws_size,
                              hipStream_t stream) {
  (void)in_sizes; (void)n_in; (void)out_size; (void)ws_size;
  const float* q  = (const float*)d_in[0];
  const float* k  = (const float*)d_in[1];
  const float* v  = (const float*)d_in[2];
  const float* wq = (const float*)d_in[3];
  const float* wk = (const float*)d_in[4];
  const float* wv = (const float*)d_in[5];
  const float* wo = (const float*)d_in[6];
  const float* bo = (const float*)d_in[7];

  float* out  = (float*)d_out;                       // [B,S,D]
  float* attn = out + (size_t)B_ * S_ * D_;          // [B,H,S,S]

  // workspace carve-up (f16 elements), total ~64 MB
  _Float16* xq16 = (_Float16*)d_ws;
  _Float16* xk16 = xq16 + (size_t)B_ * S_ * D_;
  _Float16* xv16 = xk16 + (size_t)B_ * S_ * D_;
  _Float16* wq16 = xv16 + (size_t)B_ * S_ * D_;
  _Float16* wk16 = wq16 + (size_t)D_ * D_;
  _Float16* wv16 = wk16 + (size_t)D_ * D_;
  _Float16* wo16 = wv16 + (size_t)D_ * D_;
  _Float16* qh   = wo16 + (size_t)D_ * D_;            // [B,H,S,64]
  _Float16* kh   = qh   + (size_t)B_ * H_ * S_ * DK_;
  _Float16* vt   = kh   + (size_t)B_ * H_ * S_ * DK_; // [B,H,64,S]
  _Float16* c16  = vt   + (size_t)B_ * H_ * S_ * DK_; // [B*S, D]

  const int nx = B_ * S_ * D_;
  const int nw = D_ * D_;
  mha_cvt_f32_f16<<<(nx + 255) / 256, 256, 0, stream>>>(q,  xq16, nx);
  mha_cvt_f32_f16<<<(nx + 255) / 256, 256, 0, stream>>>(k,  xk16, nx);
  mha_cvt_f32_f16<<<(nx + 255) / 256, 256, 0, stream>>>(v,  xv16, nx);
  mha_cvt_f32_f16<<<(nw + 255) / 256, 256, 0, stream>>>(wq, wq16, nw);
  mha_cvt_f32_f16<<<(nw + 255) / 256, 256, 0, stream>>>(wk, wk16, nw);
  mha_cvt_f32_f16<<<(nw + 255) / 256, 256, 0, stream>>>(wv, wv16, nw);
  mha_cvt_f32_f16<<<(nw + 255) / 256, 256, 0, stream>>>(wo, wo16, nw);

  const dim3 blk(256);
  // projections: M=4096, N=1024 -> grid 64 x 8
  mha_proj<<<dim3(64, 8), blk, 0, stream>>>(xq16, wq16, qh, 0);
  mha_proj<<<dim3(64, 8), blk, 0, stream>>>(xk16, wk16, kh, 0);
  mha_proj<<<dim3(64, 8), blk, 0, stream>>>(xv16, wv16, vt, 1);
  // fused scores+softmax+attn·V: grid = (S/16 q-strips) x (B*H)
  mha_fused_attn<<<dim3(S_ / 16, B_ * H_), blk, 0, stream>>>(qh, kh, vt, attn, c16);
  // output projection: M=4096, N=1024 -> grid 64 x 8
  mha_out<<<dim3(64, 8), blk, 0, stream>>>(c16, wo16, bo, out);
}
